// MicroController_70875550318827
// MI455X (gfx1250) — matrix operand order repaired and checked by reading, not compiled, not run on hardware
//
#include <hip/hip_runtime.h>
#include <hip/hip_bf16.h>
#include <math.h>

// MI455X (gfx1250) implementation, round 2.
// - LSTM weights packed once per launch to bf16, gate-interleaved rows (m' = 4k+gate),
//   in WMMA 16x32 A-fragment register order -> coalesced b128 A loads.
// - Gate matvec + LSTM cell nonlinearity fused in one phase: v_wmma_f32_16x16x32_bf16
//   accumulates all 4 gates of element k into one lane's registers; that lane finishes
//   sigmoid/tanh/c/h in-registers. No gates round-trip, 5 grid syncs/step (was 8).
// - B vector (4 KB) staged to LDS per phase: B reads use the LDS port, freeing the
//   VMEM pipe for the L2-resident weight stream.
// - 32 blocks x 256 threads (wave32: 8 waves/block; 256 waves == 256 gate tiles).

typedef __attribute__((ext_vector_type(16))) __bf16 v16bf;
typedef __attribute__((ext_vector_type(8)))  float  v8f;

#define NB     32
#define TPB    256
#define HDIM   1024
#define GDIM   4096      // 4*H gate outputs
#define KDIM   2048      // [x | h] concatenated
#define NTILE  256       // GDIM/16
#define NCHUNK 64        // KDIM/32
#define NSTEP  128       // action_len

// workspace float offsets
#define OFF_BAR   0                      // unsigned[2] barrier (count, gen)
#define OFF_ENT   64
#define OFF_LP    65
#define OFF_X     128                    // xcur fp32 [1024]
#define OFF_H     1152                   // h [buf*2+layer][1024] (4 slots)
#define OFF_C     (OFF_H + 4096)
#define OFF_BSUM  (OFF_C + 4096)         // permuted b_ih+b_hh [2][4096]
#define OFF_BA    (OFF_BSUM + 8192)      // = 17536 floats (32B aligned); BA[2048]bf16, BB[2048]bf16, Wpk

__device__ __forceinline__ void gsync(unsigned* bar, unsigned& gen) {
  __syncthreads();
  if (threadIdx.x == 0) {
    __threadfence();
    gen++;
    if (atomicAdd(&bar[0], 1u) == (unsigned)(NB - 1)) {
      atomicExch(&bar[0], 0u);
      __threadfence();
      atomicAdd(&bar[1], 1u);
    } else {
      while (__hip_atomic_load(&bar[1], __ATOMIC_ACQUIRE, __HIP_MEMORY_SCOPE_AGENT) < gen) {
        __builtin_amdgcn_s_sleep(2);
      }
    }
    __threadfence();
  }
  __syncthreads();
}

__device__ __forceinline__ float sigm(float x) { return 1.f / (1.f + expf(-x)); }

// One fused LSTM-layer phase: gates via WMMA + cell nonlinearity in-register.
// Each wave owns tile -> hidden elements k in [4*tile, 4*tile+4).
__device__ __forceinline__ void lstm_phase(const v16bf* __restrict__ Wl,
                                           const __hip_bfloat16* __restrict__ Bsrc,
                                           __hip_bfloat16* __restrict__ Bdst,
                                           const float* __restrict__ bsuml,
                                           const float* __restrict__ c_src,
                                           float* __restrict__ c_dst,
                                           float* __restrict__ h_dst,
                                           const float* __restrict__ other_h,
                                           bool do_write,
                                           unsigned char* __restrict__ Bsh) {
  // stage the 4 KB B vector into LDS (one b128 per thread)
  ((uint4*)Bsh)[threadIdx.x] = ((const uint4*)Bsrc)[threadIdx.x];
  __syncthreads();

  const int lane = threadIdx.x & 31;
  const int wave = threadIdx.x >> 5;
  const int tile = blockIdx.x * (TPB / 32) + wave;            // 0..255
  const unsigned char* bb = Bsh + ((lane & 16) ? 32 : 0);     // lane-half K offset (bytes)
  const v16bf* ap = Wl + (size_t)tile * NCHUNK * 32 + lane;
  v8f acc = {};
#pragma unroll 4
  for (int ch = 0; ch < NCHUNK; ++ch) {
    v16bf a = ap[(size_t)ch * 32];                            // coalesced global (L2-hot)
    v16bf b = *(const v16bf*)(bb + ch * 64);                  // LDS
    acc = __builtin_amdgcn_wmma_f32_16x16x32_bf16(false, a, false, b,
                                                  (short)0, acc, false, false);
  }
  // Column N=0 lives in lane 0 (rows 0..7) and lane 16 (rows 8..15).
  // Gate-interleaved rows: acc[4j+g] = gate g of element k = 4*tile + 2*half + j.
  if ((lane & 15) == 0) {
    const int half = lane >> 4;
    const int kb = tile * 4 + half * 2;
    const int mb = tile * 16 + half * 8;
#pragma unroll
    for (int j = 0; j < 2; ++j) {
      const int k = kb + j;
      const int m = mb + 4 * j;
      const float gi = acc[4 * j + 0] + bsuml[m + 0];
      const float gf = acc[4 * j + 1] + bsuml[m + 1];
      const float gg = acc[4 * j + 2] + bsuml[m + 2];
      const float go = acc[4 * j + 3] + bsuml[m + 3];
      const float cn = sigm(gf) * c_src[k] + sigm(gi) * tanhf(gg);
      const float hn = sigm(go) * tanhf(cn);
      c_dst[k] = cn;
      h_dst[k] = hn;
      if (do_write) {                                          // next phase's B fragment
        Bdst[k]        = __float2bfloat16(hn);
        Bdst[HDIM + k] = __float2bfloat16(other_h[k]);
      }
    }
  }
  __syncthreads();
}

__device__ __forceinline__ void build_bvec(const float* __restrict__ x,
                                           const float* __restrict__ hsrc,
                                           __hip_bfloat16* __restrict__ Bvec) {
  for (int k = threadIdx.x; k < HDIM; k += TPB) {
    Bvec[k]        = __float2bfloat16(x[k]);
    Bvec[HDIM + k] = __float2bfloat16(hsrc[k]);
  }
}

// block-0 softmax head: logits = out @ W^T + b; log-softmax; ent/lp (masked by mf)
__device__ void head_phase(const float* __restrict__ out, const float* __restrict__ W,
                           const float* __restrict__ b, int n_out, int chosen, float mf,
                           float* ent, float* lp, float* red, float* logits) {
  const int tpo = TPB / n_out;
  const int o = threadIdx.x / tpo, seg = threadIdx.x % tpo;
  const int len = HDIM / tpo;
  const float* wr = W + (size_t)o * HDIM;
  float p = 0.f;
  for (int k = seg * len; k < seg * len + len; ++k) p += out[k] * wr[k];
  red[threadIdx.x] = p;
  __syncthreads();
  if (seg == 0) {
    float s = 0.f;
    for (int j = 0; j < tpo; ++j) s += red[o * tpo + j];
    logits[o] = s + b[o];
  }
  __syncthreads();
  if (threadIdx.x == 0) {
    float mx = logits[0];
    for (int j = 1; j < n_out; ++j) mx = fmaxf(mx, logits[j]);
    float se = 0.f;
    for (int j = 0; j < n_out; ++j) se += expf(logits[j] - mx);
    float lse = mx + logf(se);
    float e = 0.f;
    for (int j = 0; j < n_out; ++j) { float ls = logits[j] - lse; e -= expf(ls) * ls; }
    *ent += mf * e;
    *lp  += mf * (logits[chosen] - lse);
  }
  __syncthreads();
}

// pack fp32 [W_ih | W_hh] -> bf16, gate-interleaved rows, WMMA A-fragment order.
// packed row m' = tile*16 + (lane&15); source row = (m'&3)*H + (m'>>2)  (gate g=m'&3, elem k=m'>>2)
// element e -> K offset: lane<16: e ; lane>=16: e+16
__global__ void pack_w(const float* __restrict__ W_ih, const float* __restrict__ W_hh,
                       __hip_bfloat16* __restrict__ Wpk) {
  size_t idx = (size_t)blockIdx.x * TPB + threadIdx.x;        // 2^24 total
  int e    = idx & 15;
  int lane = (idx >> 4) & 31;
  int ch   = (idx >> 9) & 63;
  int tile = (idx >> 15) & 255;
  int l    = (int)(idx >> 23);
  int mp   = tile * 16 + (lane & 15);
  int msrc = (mp & 3) * HDIM + (mp >> 2);
  int koff = (lane < 16) ? e : e + 16;
  int kk   = ch * 32 + koff;
  float v  = (kk < HDIM) ? W_ih[((size_t)l * GDIM + msrc) * HDIM + kk]
                         : W_hh[((size_t)l * GDIM + msrc) * HDIM + (kk - HDIM)];
  Wpk[idx] = __float2bfloat16(v);
}

__global__ void init_small(const float* __restrict__ b_ih, const float* __restrict__ b_hh,
                           float* __restrict__ F) {
  unsigned idx = blockIdx.x * TPB + threadIdx.x;
  if (idx < 2u * GDIM) {                                       // permuted bias sums
    unsigned l = idx >> 12, mp = idx & 4095u;
    unsigned msrc = l * GDIM + (mp & 3u) * HDIM + (mp >> 2);
    F[OFF_BSUM + idx] = b_ih[msrc] + b_hh[msrc];
  }
  if (idx < 64u) ((unsigned*)F)[idx] = 0u;                     // barrier region
}

__global__ void __launch_bounds__(TPB, 1)
mc_kernel(const float* __restrict__ g_emb,
          const float* __restrict__ ctx_W, const float* __restrict__ ctx_b,
          const float* __restrict__ left_W, const float* __restrict__ left_b,
          const int* __restrict__ config, const int* __restrict__ left_config,
          float* __restrict__ out, float* __restrict__ F) {
  unsigned* bar = (unsigned*)F;
  float* ent  = F + OFF_ENT;
  float* lp   = F + OFF_LP;
  float* xcur = F + OFF_X;
  float* hbuf = F + OFF_H;                   // slot (buf*2+layer)*HDIM
  float* cbuf = F + OFF_C;
  float* bsum = F + OFF_BSUM;
  __hip_bfloat16* BA = (__hip_bfloat16*)(F + OFF_BA);
  __hip_bfloat16* BB = BA + KDIM;
  const v16bf* Wpk = (const v16bf*)(BB + KDIM);
  const size_t LSTR = (size_t)NTILE * NCHUNK * 32;             // v16bf units per layer

  __shared__ __align__(64) unsigned char Bsh[4096];
  __shared__ float red[TPB];
  __shared__ float logits[32];
  unsigned gen = 0;

  if (blockIdx.x == 0) {
    for (int k = threadIdx.x; k < HDIM; k += TPB) {
      xcur[k] = g_emb[k];
      for (int s = 0; s < 4; ++s) { hbuf[s * HDIM + k] = 0.f; cbuf[s * HDIM + k] = 0.f; }
    }
    if (threadIdx.x == 0) { *ent = 0.f; *lp = 0.f; }
    __syncthreads();
    build_bvec(xcur, hbuf + 0, BA);                            // [x | h0]
  }
  gsync(bar, gen);

  for (int a = 0; a < NSTEP; ++a) {
    const int cfg = config[a];
    const int lcfg = left_config[a];

    // call 1 (buf0 -> buf1): L0 then L1
    lstm_phase(Wpk,        BA, BB, bsum,        cbuf + 0*HDIM, cbuf + 2*HDIM,
               hbuf + 2*HDIM, hbuf + 1*HDIM, true,  Bsh);      // BB = [h0_new | h1_old]
    gsync(bar, gen);
    lstm_phase(Wpk + LSTR, BB, BA, bsum + GDIM, cbuf + 1*HDIM, cbuf + 3*HDIM,
               hbuf + 3*HDIM, hbuf + 2*HDIM, true,  Bsh);      // BA = [out1 | h0_buf1]
    gsync(bar, gen);
    // call 2 / FILL (buf1 -> buf0)
    lstm_phase(Wpk,        BA, BB, bsum,        cbuf + 2*HDIM, cbuf + 0*HDIM,
               hbuf + 0*HDIM, hbuf + 3*HDIM, true,  Bsh);      // BB = [h0_new2 | h1_buf1]
    gsync(bar, gen);
    lstm_phase(Wpk + LSTR, BB, BA, bsum + GDIM, cbuf + 3*HDIM, cbuf + 1*HDIM,
               hbuf + 1*HDIM, hbuf + 0*HDIM, false, Bsh);      // out2 = h[buf0][1]
    gsync(bar, gen);

    if (blockIdx.x == 0) {                                     // heads + commit + next B
      const float* out1 = hbuf + 3 * HDIM;
      const float* out2 = hbuf + 1 * HDIM;
      const float mf = (cfg == 1) ? 1.0f : 0.0f;
      head_phase(out1, ctx_W  + (size_t)a * 16 * HDIM, ctx_b  + a * 16, 16, cfg,  1.0f,
                 ent, lp, red, logits);
      head_phase(out2, left_W + (size_t)a * 32 * HDIM, left_b + a * 32, 32, lcfg, mf,
                 ent, lp, red, logits);
      if (cfg == 1) {                                          // keep call2 state (buf0)
        for (int k = threadIdx.x; k < HDIM; k += TPB) xcur[k] = out2[k];
      } else {                                                 // revert to call1 state
        for (int k = threadIdx.x; k < HDIM; k += TPB) {
          xcur[k] = out1[k];
          hbuf[0 * HDIM + k] = hbuf[2 * HDIM + k];
          hbuf[1 * HDIM + k] = hbuf[3 * HDIM + k];
          cbuf[0 * HDIM + k] = cbuf[2 * HDIM + k];
          cbuf[1 * HDIM + k] = cbuf[3 * HDIM + k];
        }
      }
      __syncthreads();
      build_bvec(xcur, hbuf + 0, BA);                          // next step's [x | h0]
    }
    gsync(bar, gen);
  }

  if (blockIdx.x == 0 && threadIdx.x == 0) { out[0] = *ent; out[1] = *lp; }
}

extern "C" void kernel_launch(void* const* d_in, const int* in_sizes, int n_in,
                              void* d_out, int out_size, void* d_ws, size_t ws_size,
                              hipStream_t stream) {
  const float* g_emb  = (const float*)d_in[0];
  const float* W_ih   = (const float*)d_in[1];
  const float* W_hh   = (const float*)d_in[2];
  const float* b_ih   = (const float*)d_in[3];
  const float* b_hh   = (const float*)d_in[4];
  const float* ctx_W  = (const float*)d_in[5];
  const float* ctx_b  = (const float*)d_in[6];
  const float* left_W = (const float*)d_in[7];
  const float* left_b = (const float*)d_in[8];
  const int*   config = (const int*)d_in[9];
  const int*   lcfg   = (const int*)d_in[10];
  float* F = (float*)d_ws;                                     // needs ~34 MB scratch
  __hip_bfloat16* Wpk = (__hip_bfloat16*)(F + OFF_BA) + 2 * KDIM;

  hipLaunchKernelGGL(init_small, dim3(NB), dim3(TPB), 0, stream, b_ih, b_hh, F);
  hipLaunchKernelGGL(pack_w, dim3(65536), dim3(TPB), 0, stream, W_ih, W_hh, Wpk);
  hipLaunchKernelGGL(mc_kernel, dim3(NB), dim3(TPB), 0, stream,
                     g_emb, ctx_W, ctx_b, left_W, left_b, config, lcfg,
                     (float*)d_out, F);
}
